// Attention_24799141167815
// MI455X (gfx1250) — matrix operand order repaired and checked by reading, not compiled, not run on hardware
//
#include <hip/hip_runtime.h>

typedef __attribute__((ext_vector_type(16))) __bf16 v16bf;
typedef __attribute__((ext_vector_type(8)))  __bf16 v8bf;
typedef __attribute__((ext_vector_type(8)))  float  v8f;
typedef __attribute__((ext_vector_type(4)))  unsigned int v4u;
typedef __attribute__((ext_vector_type(8)))  int v8i;
typedef __attribute__((ext_vector_type(4)))  int v4i;

#define BATCH 8
#define SEQ   1024
#define CH    512
#define HEADS 8
#define HD    64
#define TROWS 3969
#define CPBH  512

static __device__ inline v8f wmma_bf16(v16bf a, v16bf b, v8f c) {
    // D = A(16x32 bf16) * B(32x16 bf16) + C(16x16 f32)
    return __builtin_amdgcn_wmma_f32_16x16x32_bf16(
        /*neg_a=*/false, a, /*neg_b=*/false, b,
        /*c_mod=*/(short)0, c, /*reuse_a=*/false, /*reuse_b=*/false);
}

static __device__ inline v16bf cat8(v8bf lo, v8bf hi) {
    return __builtin_shufflevector(lo, hi, 0, 1, 2, 3, 4, 5, 6, 7,
                                   8, 9, 10, 11, 12, 13, 14, 15);
}

// ---- A-matrix 16x32 bf16 fragment: lane = 16*kg + m.
// kg=0: elems 0-15 -> K 0-15
// kg=1: elems 0-15 -> K 16-31
static __device__ inline v16bf load_a_frag(const __bf16* row, int k0, int kg) {
    v8bf lo = *(const v8bf*)(row + k0 + kg * 16);
    v8bf hi = *(const v8bf*)(row + k0 + kg * 16 + 8);
    return cat8(lo, hi);
}
static __device__ inline v16bf load_a_frag(const float* row, int k0, int kg) {
    v8f lo = *(const v8f*)(row + k0 + kg * 16);
    v8f hi = *(const v8f*)(row + k0 + kg * 16 + 8);
    v16bf f;
#pragma unroll
    for (int e = 0; e < 8; ++e) { f[e] = (__bf16)lo[e]; f[8 + e] = (__bf16)hi[e]; }
    return f;
}

// ---- B-matrix 32x16 bf16 fragment: lane = 16*kg + n; elem e -> K = kg*16 + e.
// 'row' = length-K vector forming column n of B (row n of the row-major weight).
static __device__ inline v16bf load_b_frag(const __bf16* row, int k0, int kg) {
    return *(const v16bf*)(row + k0 + kg * 16);
}
static __device__ inline v16bf load_b_frag(const float* row, int k0, int kg) {
    v8f lo = *(const v8f*)(row + k0 + kg * 16);
    v8f hi = *(const v8f*)(row + k0 + kg * 16 + 8);
    v16bf f;
#pragma unroll
    for (int e = 0; e < 8; ++e) { f[e] = (__bf16)lo[e]; f[8 + e] = (__bf16)hi[e]; }
    return f;
}

// ---- Tensor Data Mover: DMA a (rows x 64) bf16 tile (row stride 64) -> LDS.
// D# per CDNA5 ISA ch.8: group0 = {count/lds/global/type}, group1 = dims/strides.
// This toolchain exposes the 6-arg builtin: (g0, g1, g2, g3, g_extra, cpol).
static __device__ inline void tdm_load_tile64(const __bf16* gsrc, void* ldsdst, int rows) {
    unsigned lds_off = (unsigned)(uintptr_t)ldsdst;          // flat LDS addr low 32 = offset
    unsigned long long ga = (unsigned long long)(uintptr_t)gsrc;
    v4u g0;
    g0[0] = 1u;                                              // count=1 (valid), user mode
    g0[1] = lds_off;                                         // lds_addr
    g0[2] = (unsigned)ga;                                    // global_addr[31:0]
    g0[3] = (unsigned)((ga >> 32) & 0x01FFFFFFull) | (2u << 30);  // addr[56:32] | type=2
    v8i g1;
    g1[0] = 0x00010000;                // wg_mask=0 | data_size=1 (2 bytes)
    g1[1] = (int)(64u << 16);          // tensor_dim0 = 64 (bits 79:48, low half)
    g1[2] = (int)((unsigned)SEQ << 16);// tensor_dim0 hi=0 | tensor_dim1 = 1024 (low16)
    g1[3] = (int)(64u << 16);          // tensor_dim1 hi=0 | tile_dim0 = 64
    g1[4] = rows;                      // tile_dim1 = rows | tile_dim2 = 0
    g1[5] = 64;                        // tensor_dim0_stride = 64 (low 32)
    g1[6] = 0;                         // stride0 hi | tensor_dim1_stride low16
    g1[7] = 0;
    v4i z4 = {0, 0, 0, 0};
    v8i z8 = {0, 0, 0, 0, 0, 0, 0, 0};
    __builtin_amdgcn_tensor_load_to_lds(g0, g1, z4, z4, z8, 0);
}

// ---------------------------------------------------------------------------
// Kernel 1/5: GEMM  out[M,N] = A[M,K] @ W[N,K]^T + bias   (bf16 WMMA, f32 acc)
// One wave -> one 16x64 tile (4 accumulators reuse one A fragment).
// ---------------------------------------------------------------------------
template <typename TA>
__global__ __launch_bounds__(256) void gemm16x64_kernel(
    const TA* __restrict__ A, const float* __restrict__ W,
    const float* __restrict__ bias, float* __restrict__ out,
    int M, int N, int K) {
    int wave = threadIdx.x >> 5;
    int lane = threadIdx.x & 31;
    int g  = lane >> 4;
    int ln = lane & 15;
    int nT64 = N >> 6;
    int tile = blockIdx.x * 8 + wave;
    int mT = tile / nT64;
    int n0 = (tile % nT64) << 6;

    const TA*    Arow = A + (size_t)(mT * 16 + ln) * K;
    const float* W0 = W + (size_t)(n0 + ln) * K;
    const float* W1 = W + (size_t)(n0 + 16 + ln) * K;
    const float* W2 = W + (size_t)(n0 + 32 + ln) * K;
    const float* W3 = W + (size_t)(n0 + 48 + ln) * K;

    v8f a0 = {}, a1 = {}, a2 = {}, a3 = {};
#pragma unroll 2
    for (int k0 = 0; k0 < K; k0 += 32) {
        v16bf af = load_a_frag(Arow, k0, g);
        a0 = wmma_bf16(af, load_b_frag(W0, k0, g), a0);
        a1 = wmma_bf16(af, load_b_frag(W1, k0, g), a1);
        a2 = wmma_bf16(af, load_b_frag(W2, k0, g), a2);
        a3 = wmma_bf16(af, load_b_frag(W3, k0, g), a3);
    }
    float b0 = bias[n0 + ln], b1 = bias[n0 + 16 + ln];
    float b2 = bias[n0 + 32 + ln], b3 = bias[n0 + 48 + ln];
#pragma unroll
    for (int r = 0; r < 8; ++r) {
        int row = mT * 16 + r + 8 * g;            // D layout: VGPR r -> M = r + 8*lanegroup
        float* o = out + (size_t)row * N + n0;
        o[ln]      = a0[r] + b0;
        o[16 + ln] = a1[r] + b1;
        o[32 + ln] = a2[r] + b2;
        o[48 + ln] = a3[r] + b3;
    }
}

// ---------------------------------------------------------------------------
// Kernel 2/5: normalize q/k, fold temperature scale + query embedding, pack bf16.
// One warp per (b,h,n) row-head; layout out: [b][h][n][64]
// ---------------------------------------------------------------------------
__global__ __launch_bounds__(256) void pack_qkv_kernel(
    const float* __restrict__ qkv, const float* __restrict__ temperature,
    const float* __restrict__ sls, const float* __restrict__ qe,
    __bf16* __restrict__ q_s, __bf16* __restrict__ k_n, __bf16* __restrict__ v_b) {
    int warp = (blockIdx.x * blockDim.x + threadIdx.x) >> 5;
    int lane = threadIdx.x & 31;
    int b = warp >> 13;
    int h = (warp >> 10) & 7;
    int n = warp & 1023;

    const float* row = qkv + (size_t)(b * SEQ + n) * (3 * CH);
    int d0 = lane * 2;
    float q0 = row[h * HD + d0],          q1 = row[h * HD + d0 + 1];
    float k0 = row[CH + h * HD + d0],     k1 = row[CH + h * HD + d0 + 1];
    float v0 = row[2 * CH + h * HD + d0], v1 = row[2 * CH + h * HD + d0 + 1];

    float qs = q0 * q0 + q1 * q1;
    float ks = k0 * k0 + k1 * k1;
#pragma unroll
    for (int m = 1; m <= 16; m <<= 1) {
        qs += __shfl_xor(qs, m, 32);
        ks += __shfl_xor(ks, m, 32);
    }
    float qn = fmaxf(sqrtf(qs), 1e-12f);
    float kn = fmaxf(sqrtf(ks), 1e-12f);

    float tv = temperature[h];
    float sp = (tv > 20.f) ? tv : log1pf(__expf(tv));   // softplus
    float scale = sp * sls[0];
    float e0 = qe[h * HD + d0], e1 = qe[h * HD + d0 + 1];

    size_t o = ((size_t)((b * HEADS + h) * SEQ + n)) * HD + d0;
    q_s[o]     = (__bf16)((q0 / qn + e0) * scale);
    q_s[o + 1] = (__bf16)((q1 / qn + e1) * scale);
    k_n[o]     = (__bf16)(k0 / kn);
    k_n[o + 1] = (__bf16)(k1 / kn);
    v_b[o]     = (__bf16)v0;
    v_b[o + 1] = (__bf16)v1;
}

// ---------------------------------------------------------------------------
// Kernel 3/5: CPB MLP  bias_tab[T][HEADS] = relu(coords @ W1^T + b1) @ W2^T + b2
// One warp per table row.
// ---------------------------------------------------------------------------
__global__ __launch_bounds__(256) void cpb_kernel(
    const float* __restrict__ coords, const float* __restrict__ w1,
    const float* __restrict__ b1, const float* __restrict__ w2,
    const float* __restrict__ b2, float* __restrict__ bias_tab) {
    int warp = (blockIdx.x * blockDim.x + threadIdx.x) >> 5;
    int lane = threadIdx.x & 31;
    if (warp >= TROWS) return;
    float c0 = coords[warp * 2], c1 = coords[warp * 2 + 1];
    float acc[HEADS] = {0.f, 0.f, 0.f, 0.f, 0.f, 0.f, 0.f, 0.f};
    int j0 = lane * 16;
#pragma unroll 4
    for (int j = 0; j < 16; ++j) {
        int jj = j0 + j;
        float hv = fmaxf(w1[jj * 2] * c0 + w1[jj * 2 + 1] * c1 + b1[jj], 0.f);
#pragma unroll
        for (int h = 0; h < HEADS; ++h) acc[h] += hv * w2[h * CPBH + jj];
    }
#pragma unroll
    for (int m = 1; m <= 16; m <<= 1)
#pragma unroll
        for (int h = 0; h < HEADS; ++h) acc[h] += __shfl_xor(acc[h], m, 32);
    if (lane == 0) {
#pragma unroll
        for (int h = 0; h < HEADS; ++h) bias_tab[warp * HEADS + h] = acc[h] + b2[h];
    }
}

// ---------------------------------------------------------------------------
// Kernel 4/5: flash attention per (b,h). 8 waves/block, wave = 16 q rows.
// V tiles stream in via the Tensor Data Mover (double-buffered, TENSORcnt).
// scores = q_s @ k_n^T + bias_gather ; online softmax ; O += P @ V
// out: bf16 [b][n][h][d]  (== (B,N,C) for the projection GEMM)
// ---------------------------------------------------------------------------
__global__ __launch_bounds__(256) void attn_kernel(
    const __bf16* __restrict__ q_s, const __bf16* __restrict__ k_n,
    const __bf16* __restrict__ v_b, const float* __restrict__ bias_tab,
    const int* __restrict__ rpi, __bf16* __restrict__ out) {
    __shared__ __bf16 Vstage[2][32][64]; // TDM targets, row-major (double buffer)
    __shared__ __bf16 Vt[64][40];        // V chunk transposed: Vt[d][key], padded rows
    __shared__ __bf16 Ps[8][16][32];     // per-wave P transpose staging

    int bh = blockIdx.x >> 3;            // (b*HEADS + h)
    int rb = blockIdx.x & 7;             // 128-row block
    int h  = bh & 7;
    int wave = threadIdx.x >> 5;
    int lane = threadIdx.x & 31;
    int g  = lane >> 4;
    int ln = lane & 15;
    int qrow0 = rb * 128 + wave * 16;

    const __bf16* Qrow = q_s + ((size_t)bh * SEQ + qrow0 + ln) * HD;
    v16bf qf0 = load_a_frag(Qrow, 0,  g);   // d 0..31
    v16bf qf1 = load_a_frag(Qrow, 32, g);   // d 32..63

    v8f zero = {};
    v8f O[4];
#pragma unroll
    for (int t = 0; t < 4; ++t) O[t] = zero;
    float mrow[8], lrow[8];
#pragma unroll
    for (int r = 0; r < 8; ++r) { mrow[r] = -1e30f; lrow[r] = 0.f; }

    const __bf16* Kbh = k_n + (size_t)bh * SEQ * HD;
    const __bf16* Vbh = v_b + (size_t)bh * SEQ * HD;

    const int NCH = SEQ / 32;
    if (wave == 0) tdm_load_tile64(Vbh, &Vstage[0][0][0], 32);   // prefetch chunk 0

    for (int ic = 0; ic < NCH; ++ic) {
        int kk0 = ic * 32;
        if (wave == 0) {
            if (ic + 1 < NCH) {      // issue next DMA, then wait current (in-order TDM)
                tdm_load_tile64(Vbh + (size_t)(kk0 + 32) * HD,
                                &Vstage[(ic + 1) & 1][0][0], 32);
                __builtin_amdgcn_s_wait_tensorcnt(1);
            } else {
                __builtin_amdgcn_s_wait_tensorcnt(0);
            }
        }
        __syncthreads();             // chunk ic visible to all waves; Vt safe to rebuild
        {   // transpose staged V chunk: Vt[d][key] (scatter on the store side)
            int row = threadIdx.x >> 3;           // key within chunk 0..31
            int col = (threadIdx.x & 7) * 8;      // d
            v8bf vv = *(const v8bf*)&Vstage[ic & 1][row][col];
#pragma unroll
            for (int e = 0; e < 8; ++e) Vt[col + e][row] = vv[e];
        }
        __syncthreads();

        // ---- scores: two 16x16 tiles over keys kk0..kk0+31 ----
        v8f s0 = zero, s1 = zero;
        {
            const __bf16* kr0 = Kbh + (size_t)(kk0 + ln) * HD;   // B col = key row of k_n
            s0 = wmma_bf16(qf0, load_b_frag(kr0, 0,  g), s0);
            s0 = wmma_bf16(qf1, load_b_frag(kr0, 32, g), s0);
            const __bf16* kr1 = Kbh + (size_t)(kk0 + 16 + ln) * HD;
            s1 = wmma_bf16(qf0, load_b_frag(kr1, 0,  g), s1);
            s1 = wmma_bf16(qf1, load_b_frag(kr1, 32, g), s1);
        }

        // ---- relative bias gather + online softmax ----
#pragma unroll
        for (int r = 0; r < 8; ++r) {
            int qm = qrow0 + r + 8 * g;                 // D frag: VGPR r -> row r+8g
            int t0 = rpi[(size_t)qm * SEQ + kk0 + ln];
            int t1 = rpi[(size_t)qm * SEQ + kk0 + 16 + ln];
            s0[r] += bias_tab[t0 * HEADS + h];
            s1[r] += bias_tab[t1 * HEADS + h];

            float mx = fmaxf(s0[r], s1[r]);
            mx = fmaxf(mx, __shfl_xor(mx, 1, 16));
            mx = fmaxf(mx, __shfl_xor(mx, 2, 16));
            mx = fmaxf(mx, __shfl_xor(mx, 4, 16));
            mx = fmaxf(mx, __shfl_xor(mx, 8, 16));
            float nm    = fmaxf(mrow[r], mx);
            float alpha = __expf(mrow[r] - nm);
            float p0 = __expf(s0[r] - nm);
            float p1 = __expf(s1[r] - nm);
            float ps = p0 + p1;
            ps += __shfl_xor(ps, 1, 16);
            ps += __shfl_xor(ps, 2, 16);
            ps += __shfl_xor(ps, 4, 16);
            ps += __shfl_xor(ps, 8, 16);
            lrow[r] = lrow[r] * alpha + ps;
            mrow[r] = nm;
            s0[r] = p0;
            s1[r] = p1;
#pragma unroll
            for (int t = 0; t < 4; ++t) O[t][r] *= alpha;
        }

        // ---- transpose P through per-wave LDS into an A fragment ----
#pragma unroll
        for (int r = 0; r < 8; ++r) {
            Ps[wave][r + 8 * g][ln]      = (__bf16)s0[r];
            Ps[wave][r + 8 * g][16 + ln] = (__bf16)s1[r];
        }
        asm volatile("s_wait_dscnt 0x0" ::: "memory");   // intra-wave LDS RAW
        v16bf pa;
        {
            const __bf16* pr = &Ps[wave][ln][0];
            v8bf lo = *(const v8bf*)(pr + g * 16);
            v8bf hi = *(const v8bf*)(pr + g * 16 + 8);
            pa = cat8(lo, hi);
        }

        // ---- O(16x64) += P(16x32) @ V(32x64): 4 column tiles ----
#pragma unroll
        for (int t = 0; t < 4; ++t) {
            const __bf16* vr = &Vt[t * 16 + ln][g * 16];   // B frag: contiguous keys
            v8bf lo = *(const v8bf*)vr;
            v8bf hi = *(const v8bf*)(vr + 8);
            O[t] = wmma_bf16(pa, cat8(lo, hi), O[t]);
        }
    }

    // ---- epilogue: 1/l scaling, write bf16 (b, n, h, d) ----
    int b = bh >> 3;
#pragma unroll
    for (int r = 0; r < 8; ++r) {
        float inv = 1.f / lrow[r];
        int qm = qrow0 + r + 8 * g;
        size_t base = ((size_t)(b * SEQ + qm)) * CH + h * HD;
#pragma unroll
        for (int t = 0; t < 4; ++t)
            out[base + t * 16 + ln] = (__bf16)(O[t][r] * inv);
    }
}

// ---------------------------------------------------------------------------
extern "C" void kernel_launch(void* const* d_in, const int* in_sizes, int n_in,
                              void* d_out, int out_size, void* d_ws, size_t ws_size,
                              hipStream_t stream) {
    const float* x      = (const float*)d_in[0];
    const int*   rpi    = (const int*)d_in[3];
    const float* coords = (const float*)d_in[4];
    const float* sls    = (const float*)d_in[5];
    const float* qkv_w  = (const float*)d_in[6];
    const float* qkv_b  = (const float*)d_in[7];
    const float* proj_w = (const float*)d_in[8];
    const float* proj_b = (const float*)d_in[9];
    const float* temp   = (const float*)d_in[10];
    const float* qe     = (const float*)d_in[11];
    const float* fc1w   = (const float*)d_in[12];
    const float* fc1b   = (const float*)d_in[13];
    const float* fc2w   = (const float*)d_in[14];
    const float* fc2b   = (const float*)d_in[15];

    char* wsb = (char*)d_ws;
    size_t off = 0;
    auto carve = [&](size_t bytes) -> void* {
        void* p = wsb + off;
        off = (off + bytes + 255) & ~(size_t)255;
        return p;
    };
    float*  qkv      = (float*)carve((size_t)BATCH * SEQ * 3 * CH * sizeof(float));
    __bf16* q_s      = (__bf16*)carve((size_t)BATCH * HEADS * SEQ * HD * sizeof(__bf16));
    __bf16* k_n      = (__bf16*)carve((size_t)BATCH * HEADS * SEQ * HD * sizeof(__bf16));
    __bf16* v_b      = (__bf16*)carve((size_t)BATCH * HEADS * SEQ * HD * sizeof(__bf16));
    float*  bias_tab = (float*)carve((size_t)TROWS * HEADS * sizeof(float));
    __bf16* attn_out = (__bf16*)carve((size_t)BATCH * SEQ * CH * sizeof(__bf16));

    const int M = BATCH * SEQ;  // 8192

    // 1) qkv = x @ qkv_w^T + qkv_b   (8192x512 @ 512x1536)
    {
        int tiles = (M / 16) * ((3 * CH) / 64);   // 512 * 24
        gemm16x64_kernel<float><<<tiles / 8, 256, 0, stream>>>(
            x, qkv_w, qkv_b, qkv, M, 3 * CH, CH);
    }
    // 2) l2norm + embed + scale -> bf16 q_s/k_n/v_b
    {
        int warps = BATCH * HEADS * SEQ;  // 65536
        pack_qkv_kernel<<<warps / 8, 256, 0, stream>>>(qkv, temp, sls, qe, q_s, k_n, v_b);
    }
    // 3) CPB MLP -> bias_tab[T][HEADS]
    {
        int blocks = (TROWS + 7) / 8;
        cpb_kernel<<<blocks, 256, 0, stream>>>(coords, fc1w, fc1b, fc2w, fc2b, bias_tab);
    }
    // 4) flash attention (TDM-fed) -> attn_out bf16 (B,N,C)
    {
        attn_kernel<<<BATCH * HEADS * (SEQ / 128), 256, 0, stream>>>(
            q_s, k_n, v_b, bias_tab, rpi, attn_out);
    }
    // 5) out = attn_out @ proj_w^T + proj_b  (fp32 result)
    {
        int tiles = (M / 16) * (CH / 64);         // 512 * 8
        gemm16x64_kernel<__bf16><<<tiles / 8, 256, 0, stream>>>(
            attn_out, proj_w, proj_b, (float*)d_out, M, CH, CH);
    }
}
